// SimpleMPONetwork_7284264534631
// MI455X (gfx1250) — compile-verified, matrix operand order checked by reading, not run on hardware
//
#include <hip/hip_runtime.h>
#include <hip/hip_bf16.h>
#include <stddef.h>

// ---------------------------------------------------------------------------
// MPO network on gfx1250 (MI455X, wave32).
// Big GEMMs run on v_wmma_f32_16x16x32_bf16 using a hi/lo bf16 split of the
// fp32 operands (3 WMMAs per product -> ~fp32 accuracy at bf16-pipe rate).
// GEMM uses double-buffered LDS tiles: one barrier per K-slab, next slab's
// global loads issued before the WMMA block so HBM/L2 latency is hidden.
// TT-SVD is done as: M = A A^T (WMMA GEMM), top-2 eigenvectors via f64
// simultaneous power iteration (one workgroup), T = U^T A.
// ---------------------------------------------------------------------------

typedef __bf16 bf16_t;
typedef __attribute__((ext_vector_type(16))) __bf16 v16bf;
typedef __attribute__((ext_vector_type(8)))  __bf16 v8bf;
typedef __attribute__((ext_vector_type(8)))  float  v8f;

#define LDSK       32   // K-slab per iteration (one bf16 WMMA depth)
#define LDS_STRIDE 40   // bf16 elems per LDS row: 32 + 8 pad = 80 B
                        // (16B-aligned rows; 20-bank stride -> conflict-free b128 reads)

__device__ __forceinline__ void cvt_split_store(bf16_t* hi, bf16_t* lo, int idx, float x) {
  bf16_t h = (bf16_t)x;                 // RNE truncate to bf16
  hi[idx] = h;
  lo[idx] = (bf16_t)(x - (float)h);     // residual
}

__device__ __forceinline__ void split_store4(bf16_t* hi, bf16_t* lo, int idx,
                                             float4 v, int relu) {
  if (relu) {
    v.x = fmaxf(v.x, 0.f); v.y = fmaxf(v.y, 0.f);
    v.z = fmaxf(v.z, 0.f); v.w = fmaxf(v.w, 0.f);
  }
  cvt_split_store(hi, lo, idx + 0, v.x);
  cvt_split_store(hi, lo, idx + 1, v.y);
  cvt_split_store(hi, lo, idx + 2, v.z);
  cvt_split_store(hi, lo, idx + 3, v.w);
}

struct F16R { float4 v[4]; };           // one thread's 16-float slab row chunk

__device__ __forceinline__ F16R load16(const float* p) {
  F16R r;
  const float4* q = (const float4*)p;
  r.v[0] = q[0]; r.v[1] = q[1]; r.v[2] = q[2]; r.v[3] = q[3];
  return r;
}

// Load one 16x32 bf16 fragment (A-layout per ISA 7.12.2; B assumed symmetric
// over columns, tile stored N-major x K in LDS so the same loader applies).
// lanes 0-15 : row = rowBase+lane, K 0..7 -> VGPR0-3, K 16..23 -> VGPR4-7
// lanes16-31 : row = rowBase+lane-16, K 8..15 / 24..31
__device__ __forceinline__ v16bf load_frag(const bf16_t* smem, int rowBase, int lane) {
  int r  = rowBase + (lane & 15);
  int kh = (lane >> 4) << 3;            // 0 or 8 elements
  const bf16_t* p = smem + r * LDS_STRIDE;
  union { v16bf v; struct { v8bf a, b; } s; } u;
  u.s.a = *(const v8bf*)(p + kh);       // ds_read_b128
  u.s.b = *(const v8bf*)(p + 16 + kh);  // ds_read_b128
  return u.v;
}

// C = [reluOut]( A @ B^T + bias ), A:(M,K) B:(N,K) C:(M,N) all row-major f32.
// reluA applies relu to A on load. Requires M%128==0, N%128==0, K%32==0.
__global__ __launch_bounds__(256) void wmma_gemm_split(
    const float* __restrict__ A, const float* __restrict__ B,
    float* __restrict__ C, const float* __restrict__ bias,
    int M, int N, int K, int reluA, int reluOut)
{
  __shared__ bf16_t sAhi[2][128 * LDS_STRIDE];
  __shared__ bf16_t sAlo[2][128 * LDS_STRIDE];
  __shared__ bf16_t sBhi[2][128 * LDS_STRIDE];
  __shared__ bf16_t sBlo[2][128 * LDS_STRIDE];

  const int t     = threadIdx.x;
  const int lane  = t & 31;
  const int wave  = t >> 5;
  const int wm    = wave >> 2;          // 0..1 : 64-row band
  const int wn    = wave & 3;           // 0..3 : 32-col band
  const int mBase = blockIdx.y * 128;
  const int nBase = blockIdx.x * 128;

  const int lr = t >> 1;                // 0..127 : tile row this thread loads
  const int lc = (t & 1) << 4;          // 0 / 16 : starting K within slab

  const float* gA = A + (size_t)(mBase + lr) * K + lc;
  const float* gB = B + (size_t)(nBase + lr) * K + lc;

  v8f acc[4][2];
#pragma unroll
  for (int mt = 0; mt < 4; ++mt)
#pragma unroll
    for (int nt = 0; nt < 2; ++nt)
#pragma unroll
      for (int e = 0; e < 8; ++e) acc[mt][nt][e] = 0.0f;

  // prologue: fetch slab 0 into registers
  F16R ra = load16(gA);
  F16R rb = load16(gB);
  int buf = 0;

  for (int k0 = 0; k0 < K; k0 += LDSK) {
    // stage current slab registers -> LDS[buf] with hi/lo bf16 split
    const int base = lr * LDS_STRIDE + lc;
    split_store4(sAhi[buf], sAlo[buf], base +  0, ra.v[0], reluA);
    split_store4(sAhi[buf], sAlo[buf], base +  4, ra.v[1], reluA);
    split_store4(sAhi[buf], sAlo[buf], base +  8, ra.v[2], reluA);
    split_store4(sAhi[buf], sAlo[buf], base + 12, ra.v[3], reluA);
    split_store4(sBhi[buf], sBlo[buf], base +  0, rb.v[0], 0);
    split_store4(sBhi[buf], sBlo[buf], base +  4, rb.v[1], 0);
    split_store4(sBhi[buf], sBlo[buf], base +  8, rb.v[2], 0);
    split_store4(sBhi[buf], sBlo[buf], base + 12, rb.v[3], 0);
    __syncthreads();                    // single barrier per slab (double buffer)

    // issue next slab's global loads; latency covered by the WMMA block below
    if (k0 + LDSK < K) {
      ra = load16(gA + k0 + LDSK);
      rb = load16(gB + k0 + LDSK);
      if (k0 + 2 * LDSK < K) {          // global_prefetch_b8 two slabs ahead
        __builtin_prefetch(gA + k0 + 2 * LDSK, 0, 1);
        __builtin_prefetch(gB + k0 + 2 * LDSK, 0, 1);
      }
    }

    v16bf aHi[4], aLo[4], bHi[2], bLo[2];
#pragma unroll
    for (int mt = 0; mt < 4; ++mt) {
      int rbse = wm * 64 + mt * 16;
      aHi[mt] = load_frag(sAhi[buf], rbse, lane);
      aLo[mt] = load_frag(sAlo[buf], rbse, lane);
    }
#pragma unroll
    for (int nt = 0; nt < 2; ++nt) {
      int rbse = wn * 32 + nt * 16;
      bHi[nt] = load_frag(sBhi[buf], rbse, lane);
      bLo[nt] = load_frag(sBlo[buf], rbse, lane);
    }
#pragma unroll
    for (int mt = 0; mt < 4; ++mt)
#pragma unroll
      for (int nt = 0; nt < 2; ++nt) {
        v8f c = acc[mt][nt];
        c = __builtin_amdgcn_wmma_f32_16x16x32_bf16(false, aHi[mt], false, bHi[nt],
                                                    (short)0, c, false, false);
        c = __builtin_amdgcn_wmma_f32_16x16x32_bf16(false, aHi[mt], false, bLo[nt],
                                                    (short)0, c, false, false);
        c = __builtin_amdgcn_wmma_f32_16x16x32_bf16(false, aLo[mt], false, bHi[nt],
                                                    (short)0, c, false, false);
        acc[mt][nt] = c;
      }
    buf ^= 1;
  }

  // C layout: VGPR r -> (M = r + 8*(lane>>4), N = lane&15) within 16x16 tile
  const int cn  = lane & 15;
  const int mh8 = (lane >> 4) << 3;
#pragma unroll
  for (int mt = 0; mt < 4; ++mt)
#pragma unroll
    for (int nt = 0; nt < 2; ++nt) {
      int row0 = mBase + wm * 64 + mt * 16 + mh8;
      int col  = nBase + wn * 32 + nt * 16 + cn;
      float bv = bias ? bias[col] : 0.0f;
#pragma unroll
      for (int r = 0; r < 8; ++r) {
        float v = acc[mt][nt][r] + bv;
        if (reluOut) v = fmaxf(v, 0.f);
        C[(size_t)(row0 + r) * N + col] = v;
      }
    }
}

// ---------------------------------------------------------------------------
// A1[p,q] = W_mpo interleave-permuted unfolding: p=(o1,i1), q=(o2,i2,o3,i3)
__global__ __launch_bounds__(256) void pack_A1(const float* __restrict__ W,
                                               float* __restrict__ A1)
{
  int idx = blockIdx.x * 256 + threadIdx.x;   // 16,777,216 total
  int p = idx >> 16, q = idx & 0xFFFF;
  int o1 = p >> 4, i1 = p & 15;
  int o2 = q >> 12, i2 = (q >> 8) & 15, o3 = (q >> 4) & 15, i3 = q & 15;
  int row = (o1 << 8) | (o2 << 4) | o3;
  int col = (i1 << 8) | (i2 << 4) | i3;
  A1[idx] = W[(size_t)row * 4096 + col];
}

// ---------------------------------------------------------------------------
__device__ __forceinline__ double block_sum(double x, double* red) {
  int t = threadIdx.x;
  red[t] = x; __syncthreads();
  for (int s = 128; s > 0; s >>= 1) {
    if (t < s) red[t] += red[t + s];
    __syncthreads();
  }
  double r = red[0]; __syncthreads();
  return r;
}

// Top-2 eigenvectors of SPD Mm (n x n, n in {256,512}) by f64 simultaneous
// power iteration with Gram-Schmidt. One workgroup, deterministic init.
__global__ __launch_bounds__(256) void top2_eig(const float* __restrict__ Mm,
                                                int n, float* __restrict__ U,
                                                int iters)
{
  __shared__ double v0[512], v1[512], w0[512], w1[512];
  __shared__ double red[256];
  const int t = threadIdx.x;
  const int rows = n >> 8;              // 1 or 2 rows per thread

  for (int i = t; i < n; i += 256) {
    unsigned a = (unsigned)i * 2654435761u ^ 0x9E3779B9u;
    a ^= a >> 13; a *= 0x85EBCA6Bu; a ^= a >> 16;
    unsigned b = (unsigned)(i + 77) * 2246822519u ^ 0xB5297A4Du;
    b ^= b >> 15; b *= 0xC2B2AE35u; b ^= b >> 13;
    v0[i] = (double)(a & 0xFFFF) / 65536.0 + 0.05;
    v1[i] = (double)(b & 0xFFFF) / 65536.0 - 0.5;
  }
  __syncthreads();

  for (int it = 0; it < iters; ++it) {
    for (int rr = 0; rr < rows; ++rr) {
      int i = t + (rr << 8);
      const float* row = Mm + (size_t)i * n;
      double a0 = 0.0, a1 = 0.0;
      for (int j = 0; j < n; ++j) {
        double m = (double)row[j];
        a0 += m * v0[j]; a1 += m * v1[j];
      }
      w0[i] = a0; w1[i] = a1;
    }
    __syncthreads();

    double p0 = 0.0;
    for (int rr = 0; rr < rows; ++rr) { int i = t + (rr << 8); p0 += w0[i] * w0[i]; }
    double inv0 = 1.0 / sqrt(block_sum(p0, red) + 1e-300);
    for (int rr = 0; rr < rows; ++rr) { int i = t + (rr << 8); w0[i] *= inv0; }
    __syncthreads();

    double pd = 0.0;
    for (int rr = 0; rr < rows; ++rr) { int i = t + (rr << 8); pd += w0[i] * w1[i]; }
    double d01 = block_sum(pd, red);
    for (int rr = 0; rr < rows; ++rr) { int i = t + (rr << 8); w1[i] -= d01 * w0[i]; }
    __syncthreads();

    double p1 = 0.0;
    for (int rr = 0; rr < rows; ++rr) { int i = t + (rr << 8); p1 += w1[i] * w1[i]; }
    double inv1 = 1.0 / sqrt(block_sum(p1, red) + 1e-300);
    for (int rr = 0; rr < rows; ++rr) {
      int i = t + (rr << 8);
      v0[i] = w0[i]; v1[i] = w1[i] * inv1;
    }
    __syncthreads();
  }

  for (int i = t; i < n; i += 256) {
    U[i * 2 + 0] = (float)v0[i];
    U[i * 2 + 1] = (float)v1[i];
  }
}

// T(2,qdim) = U^T(2,pdim) @ A(pdim,qdim)
__global__ __launch_bounds__(256) void project_rows(
    const float* __restrict__ A, const float* __restrict__ U,
    float* __restrict__ T, int pdim, int qdim)
{
  __shared__ float su0[512], su1[512];
  const int t = threadIdx.x;
  for (int i = t; i < pdim; i += 256) { su0[i] = U[i * 2]; su1[i] = U[i * 2 + 1]; }
  __syncthreads();
  int q = blockIdx.x * 256 + t;
  if (q < qdim) {
    float a0 = 0.f, a1 = 0.f;
    for (int p = 0; p < pdim; ++p) {
      float x = A[(size_t)p * qdim + q];
      a0 += su0[p] * x; a1 += su1[p] * x;
    }
    T[q] = a0; T[qdim + q] = a1;
  }
}

// ---------------------------------------------------------------------------
// Per-sample TT matvec, in place on h:  y[I,K,M] = sum G1[I,J,a] G2[a,K,L,b]
// G3[b,M,N] h[J,L,N], then h <- y + b_mpo (relu deferred to GEMM3 A-load).
// G1 = U1 flat (256,2); G2 = U2 flat (512,2); G3 = T3 flat (2,256).
__global__ __launch_bounds__(256) void tt_contract(
    float* __restrict__ h, const float* __restrict__ G1,
    const float* __restrict__ G2, const float* __restrict__ G3,
    const float* __restrict__ bmpo)
{
  __shared__ float sh[4096];
  __shared__ float t1[4096];            // [J,L,b,Mr] for one M-half (Mr 0..7)
  __shared__ float t2[4096];            // [a,K,J,Mr]
  __shared__ float sg1[512], sg2[1024], sg3[512];
  const int t = threadIdx.x;
  const size_t base = (size_t)blockIdx.x * 4096;

  for (int i = t; i < 4096; i += 256) sh[i]  = h[base + i];
  for (int i = t; i < 512;  i += 256) sg1[i] = G1[i];
  for (int i = t; i < 1024; i += 256) sg2[i] = G2[i];
  for (int i = t; i < 512;  i += 256) sg3[i] = G3[i];
  __syncthreads();

  for (int mh = 0; mh < 2; ++mh) {
    const int Mb = mh * 8;
    // t1[J,L,b,Mr] = sum_N G3[b, Mb+Mr, N] * h[J,L,N]
    for (int o = t; o < 4096; o += 256) {
      int J = o >> 8, L = (o >> 4) & 15, b2 = (o >> 3) & 1, Mr = o & 7;
      const float* g3 = sg3 + b2 * 256 + (Mb + Mr) * 16;
      const float* hh = sh + J * 256 + L * 16;
      float a = 0.f;
#pragma unroll
      for (int N = 0; N < 16; ++N) a += g3[N] * hh[N];
      t1[o] = a;
    }
    __syncthreads();
    // t2[a,K,J,Mr] = sum_{L,b} G2[a,K,L,b] * t1[J,L,b,Mr]
    for (int o = t; o < 4096; o += 256) {
      int a_ = o >> 11, K = (o >> 7) & 15, J = (o >> 3) & 15, Mr = o & 7;
      const float* g2 = sg2 + (a_ * 256 + K * 16) * 2;
      float acc = 0.f;
#pragma unroll
      for (int L = 0; L < 16; ++L) {
        acc += g2[L * 2 + 0] * t1[((J * 16 + L) * 2 + 0) * 8 + Mr];
        acc += g2[L * 2 + 1] * t1[((J * 16 + L) * 2 + 1) * 8 + Mr];
      }
      t2[o] = acc;
    }
    __syncthreads();
    // y[I,K,M] = sum_{J,a} G1[I,J,a] * t2[a,K,J,Mr]
    for (int o = t; o < 2048; o += 256) {
      int I = o >> 7, K = (o >> 3) & 15, Mr = o & 7;
      const float* g1 = sg1 + I * 32;
      float acc = 0.f;
#pragma unroll
      for (int J = 0; J < 16; ++J) {
        acc += g1[J * 2 + 0] * t2[((0 * 16 + K) * 16 + J) * 8 + Mr];
        acc += g1[J * 2 + 1] * t2[((1 * 16 + K) * 16 + J) * 8 + Mr];
      }
      int oi = (I * 16 + K) * 16 + (Mb + Mr);
      h[base + oi] = acc + bmpo[oi];
    }
    __syncthreads();
  }
}

// ---------------------------------------------------------------------------
extern "C" void kernel_launch(void* const* d_in, const int* in_sizes, int n_in,
                              void* d_out, int out_size, void* d_ws, size_t ws_size,
                              hipStream_t stream) {
  (void)in_sizes; (void)n_in; (void)out_size; (void)ws_size;
  const float* x     = (const float*)d_in[0];   // (8192,2048)
  const float* W_in  = (const float*)d_in[1];   // (4096,2048)
  const float* b_in  = (const float*)d_in[2];   // (4096)
  const float* W_mpo = (const float*)d_in[3];   // (4096,4096)
  const float* b_mpo = (const float*)d_in[4];   // (4096)
  const float* W_out = (const float*)d_in[5];   // (1024,4096)
  const float* b_out = (const float*)d_in[6];   // (1024)
  float* out = (float*)d_out;                   // (8192,1024)

  // workspace layout (floats); peak ~204 MB
  float* ws = (float*)d_ws;
  float* h  = ws;                                    // 8192*4096
  float* A1 = h  + (size_t)8192 * 4096;              // 256*65536
  float* M1 = A1 + (size_t)256 * 65536;              // 256*256
  float* U1 = M1 + 65536;                            // 256*2   (== G1)
  float* T2 = U1 + 512;                              // 2*65536 (reshape-aliases A2 512x256)
  float* A2 = T2;
  float* M2 = T2 + 131072;                           // 512*512
  float* U2 = M2 + 262144;                           // 512*2   (== G2)
  float* T3 = U2 + 1024;                             // 2*256   (== G3)

  // --- TT-SVD of W_mpo -----------------------------------------------------
  pack_A1<<<dim3(65536), dim3(256), 0, stream>>>(W_mpo, A1);
  // M1 = A1 @ A1^T  (256x256, K=65536)
  wmma_gemm_split<<<dim3(2, 2), dim3(256), 0, stream>>>(
      A1, A1, M1, nullptr, 256, 256, 65536, 0, 0);
  top2_eig<<<dim3(1), dim3(256), 0, stream>>>(M1, 256, U1, 512);
  project_rows<<<dim3(256), dim3(256), 0, stream>>>(A1, U1, T2, 256, 65536);
  // M2 = A2 @ A2^T  (512x512, K=256); A2 = reshape(T2, 512, 256) is an alias
  wmma_gemm_split<<<dim3(4, 4), dim3(256), 0, stream>>>(
      A2, A2, M2, nullptr, 512, 512, 256, 0, 0);
  top2_eig<<<dim3(1), dim3(256), 0, stream>>>(M2, 512, U2, 512);
  project_rows<<<dim3(1), dim3(256), 0, stream>>>(A2, U2, T3, 512, 256);

  // --- forward pass --------------------------------------------------------
  // h = relu(x @ W_in^T + b_in)
  wmma_gemm_split<<<dim3(4096 / 128, 8192 / 128), dim3(256), 0, stream>>>(
      x, W_in, h, b_in, 8192, 4096, 2048, 0, 1);
  // h <- TT(h) + b_mpo (per-sample contraction, in place)
  tt_contract<<<dim3(8192), dim3(256), 0, stream>>>(h, U1, U2, T3, b_mpo);
  // out = relu(h) @ W_out^T + b_out (relu fused into A-load)
  wmma_gemm_split<<<dim3(1024 / 128, 8192 / 128), dim3(256), 0, stream>>>(
      h, W_out, out, b_out, 8192, 1024, 4096, 1, 0);
}
